// ScaledDotProductAttention_28595892257161
// MI455X (gfx1250) — matrix operand order repaired and checked by reading, not compiled, not run on hardware
//
#include <hip/hip_runtime.h>
#include <hip/hip_bf16.h>

typedef __attribute__((ext_vector_type(16))) _Float16 v16h;
typedef __attribute__((ext_vector_type(8)))  _Float16 v8h;
typedef __attribute__((ext_vector_type(8)))  float    v8f;
typedef __attribute__((ext_vector_type(4)))  float    v4f;

#define SEQ     2048
#define HD      64
#define TILE_Q  64
#define TILE_K  64
#define NTHREAD 128
#define NEG_INF (-1.0e9f)
#define INV_T   0.125f   // 1/temperature, temperature = 8 = sqrt(64)

// ---- fragment loaders (wave32, 16-bit WMMA layouts per CDNA5 ISA 7.12.2) ----

// A-matrix 16x32 f16: lane m=l&15 group g=l>>4; VGPR0-3 hold K = g*8+[0,8),
// VGPR4-7 hold K = 16+g*8+[0,8). Two contiguous 16B chunks per lane.
__device__ inline v16h load_frag_a(const _Float16* __restrict__ src, int stride,
                                   int cb, int lane) {
    const int m = lane & 15, g = lane >> 4;
    const _Float16* p = src + m * stride + cb + g * 8;
    v8h lo = *(const v8h*)p;
    v8h hi = *(const v8h*)(p + 16);
    v16h a;
#pragma unroll
    for (int i = 0; i < 8; ++i) { a[i] = lo[i]; a[8 + i] = hi[i]; }
    return a;
}

// B-matrix 32x16 f16, loaded from B^T stored row-major (srcT[n][k]):
// lane n=l&15 group g=l>>4; VGPR0-3 hold K = kb+g*8+[0,8),
// VGPR4-7 hold K = kb+16+g*8+[0,8). Two contiguous 16B chunks per lane.
__device__ inline v16h load_frag_b(const _Float16* __restrict__ srcT, int stride,
                                   int nb, int kb, int lane) {
    const int n = lane & 15, g = lane >> 4;
    const _Float16* p = srcT + (nb + n) * stride + kb + g * 8;
    v8h lo = *(const v8h*)p;
    v8h hi = *(const v8h*)(p + 16);
    v16h b;
#pragma unroll
    for (int i = 0; i < 8; ++i) { b[i] = lo[i]; b[8 + i] = hi[i]; }
    return b;
}

__device__ inline v8f wmma_f16(v16h a, v16h b, v8f c) {
    return __builtin_amdgcn_wmma_f32_16x16x32_f16(false, a, false, b,
                                                  (short)0, c, false, false);
}

__device__ inline float half_reduce_max(float x) {
    x = fmaxf(x, __shfl_xor(x, 1, 32));
    x = fmaxf(x, __shfl_xor(x, 2, 32));
    x = fmaxf(x, __shfl_xor(x, 4, 32));
    x = fmaxf(x, __shfl_xor(x, 8, 32));
    return x;
}
__device__ inline float half_reduce_sum(float x) {
    x += __shfl_xor(x, 1, 32);
    x += __shfl_xor(x, 2, 32);
    x += __shfl_xor(x, 4, 32);
    x += __shfl_xor(x, 8, 32);
    return x;
}

__global__ __launch_bounds__(NTHREAD)
void ScaledDotProductAttention_28595892257161_kernel(
        const float* __restrict__ q, const float* __restrict__ k,
        const float* __restrict__ v, const int* __restrict__ mask,
        float* __restrict__ out, float* __restrict__ attn)
{
    __shared__ _Float16 Qs[TILE_Q * HD];        // [qrow][d], pre-scaled by 1/T
    __shared__ _Float16 Ks[TILE_K * HD];        // [key][d]  (== (K^T)^T rows)
    __shared__ _Float16 Vt[HD * TILE_K];        // [d][key]  (V transposed)
    __shared__ _Float16 Ps[4][16 * TILE_K];     // per-wave P tile [row16][key64]

    const int qtile = blockIdx.x;               // 0..S/64-1
    const int bh    = blockIdx.y;               // 0..B*H-1
    const int tid   = threadIdx.x;
    const int lane  = tid & 31;
    const int wave  = tid >> 5;
    const int g     = lane >> 4;                // C-layout lane group
    const int n     = lane & 15;                // C-layout column within tile

    const float* qp = q + ((size_t)bh * SEQ + (size_t)qtile * TILE_Q) * HD;
    const float* kp = k + (size_t)bh * SEQ * HD;
    const float* vp = v + (size_t)bh * SEQ * HD;
    float* attn_b   = attn + (size_t)bh * SEQ * SEQ;

    // ---- stage Q once (scaled by 1/T), f32 -> f16 ----
#pragma unroll
    for (int i = tid; i < (TILE_Q * HD) / 4; i += NTHREAD) {
        v4f x = *(const v4f*)(qp + 4 * i);
        _Float16* d = Qs + 4 * i;
        d[0] = (_Float16)(x[0] * INV_T); d[1] = (_Float16)(x[1] * INV_T);
        d[2] = (_Float16)(x[2] * INV_T); d[3] = (_Float16)(x[3] * INV_T);
    }
    __syncthreads();

    // persistent Q A-fragments for this wave's 16 rows (K=64 -> two K=32 slabs)
    const v16h qa0 = load_frag_a(Qs + wave * 16 * HD, HD, 0,  lane);
    const v16h qa1 = load_frag_a(Qs + wave * 16 * HD, HD, 32, lane);

    const int qrow0 = qtile * TILE_Q + wave * 16 + 8 * g;  // row for reg r: qrow0 + r

    float mrow[8], lrow[8];
#pragma unroll
    for (int r = 0; r < 8; ++r) { mrow[r] = NEG_INF; lrow[r] = 0.0f; }

    // =========================== pass 1: row max & sum ===========================
    for (int kc = 0; kc < SEQ / TILE_K; ++kc) {
        __syncthreads();
        const float* kt = kp + (size_t)kc * TILE_K * HD;
#pragma unroll
        for (int i = tid; i < (TILE_K * HD) / 4; i += NTHREAD) {
            v4f x = *(const v4f*)(kt + 4 * i);
            _Float16* d = Ks + 4 * i;
            d[0] = (_Float16)x[0]; d[1] = (_Float16)x[1];
            d[2] = (_Float16)x[2]; d[3] = (_Float16)x[3];
            if (kc + 1 < SEQ / TILE_K)
                __builtin_prefetch(kt + TILE_K * HD + 4 * i, 0, 1);
        }
        __syncthreads();

        v8f s[4];
#pragma unroll
        for (int t = 0; t < 4; ++t) {
            v16h b0 = load_frag_b(Ks, HD, t * 16, 0,  lane);
            v16h b1 = load_frag_b(Ks, HD, t * 16, 32, lane);
            v8f c = {};
            c = wmma_f16(qa0, b0, c);
            c = wmma_f16(qa1, b1, c);
            s[t] = c;
        }
        // abs + mask
#pragma unroll
        for (int t = 0; t < 4; ++t) {
            const int key = kc * TILE_K + t * 16 + n;
#pragma unroll
            for (int r = 0; r < 8; ++r) {
                float val = fabsf(s[t][r]);
                if (mask[(size_t)(qrow0 + r) * SEQ + key] == 0) val = NEG_INF;
                s[t][r] = val;
            }
        }
        // online max / sum per row
#pragma unroll
        for (int r = 0; r < 8; ++r) {
            float x = fmaxf(fmaxf(s[0][r], s[1][r]), fmaxf(s[2][r], s[3][r]));
            x = half_reduce_max(x);
            const float nm    = fmaxf(mrow[r], x);
            const float scale = __expf(mrow[r] - nm);
            float sum = __expf(s[0][r] - nm) + __expf(s[1][r] - nm) +
                        __expf(s[2][r] - nm) + __expf(s[3][r] - nm);
            sum = half_reduce_sum(sum);
            lrow[r] = lrow[r] * scale + sum;
            mrow[r] = nm;
        }
    }

    float linv[8];
#pragma unroll
    for (int r = 0; r < 8; ++r) linv[r] = (lrow[r] > 0.0f) ? (1.0f / lrow[r]) : 0.0f;

    // ================= pass 2: recompute, store attn, accumulate P@V =================
    v8f acc[4] = {};                            // out tile 16x64 (4 N-tiles of 16)
    _Float16* myP = Ps[wave];

    for (int kc = 0; kc < SEQ / TILE_K; ++kc) {
        __syncthreads();
        const float* kt = kp + (size_t)kc * TILE_K * HD;
        const float* vt = vp + (size_t)kc * TILE_K * HD;
#pragma unroll
        for (int i = tid; i < (TILE_K * HD) / 4; i += NTHREAD) {
            v4f x = *(const v4f*)(kt + 4 * i);
            _Float16* d = Ks + 4 * i;
            d[0] = (_Float16)x[0]; d[1] = (_Float16)x[1];
            d[2] = (_Float16)x[2]; d[3] = (_Float16)x[3];
            // V: transpose into Vt[d][key]
            v4f y = *(const v4f*)(vt + 4 * i);
            const int key = (4 * i) / HD, dd = (4 * i) % HD;
            Vt[(dd + 0) * TILE_K + key] = (_Float16)y[0];
            Vt[(dd + 1) * TILE_K + key] = (_Float16)y[1];
            Vt[(dd + 2) * TILE_K + key] = (_Float16)y[2];
            Vt[(dd + 3) * TILE_K + key] = (_Float16)y[3];
        }
        __syncthreads();

        // scores -> probabilities; store attn + stage P (one ctile at a time
        // to keep register pressure low)
#pragma unroll
        for (int t = 0; t < 4; ++t) {
            v16h b0 = load_frag_b(Ks, HD, t * 16, 0,  lane);
            v16h b1 = load_frag_b(Ks, HD, t * 16, 32, lane);
            v8f c = {};
            c = wmma_f16(qa0, b0, c);
            c = wmma_f16(qa1, b1, c);
            const int key = kc * TILE_K + t * 16 + n;
#pragma unroll
            for (int r = 0; r < 8; ++r) {
                float val = fabsf(c[r]);
                if (mask[(size_t)(qrow0 + r) * SEQ + key] == 0) val = NEG_INF;
                const float p = __expf(val - mrow[r]) * linv[r];
                attn_b[(size_t)(qrow0 + r) * SEQ + key] = p;
                myP[(r + 8 * g) * TILE_K + t * 16 + n] = (_Float16)p;
            }
        }
        // wave-local LDS RAW: P stores above -> A-fragment loads below
        asm volatile("s_wait_dscnt 0" ::: "memory");

        // out += P @ V   (K = 64 keys -> two K=32 slabs, 4 N-tiles over d)
#pragma unroll
        for (int kk = 0; kk < 2; ++kk) {
            v16h pa = load_frag_a(myP, TILE_K, kk * 32, lane);
#pragma unroll
            for (int t = 0; t < 4; ++t) {
                v16h vb = load_frag_b(Vt, TILE_K, t * 16, kk * 32, lane);
                acc[t] = wmma_f16(pa, vb, acc[t]);
            }
        }
    }

    // ---- write out tile: row M = r+8g, d = t*16+n ----
    float* ob = out + ((size_t)bh * SEQ + (size_t)qtile * TILE_Q + wave * 16) * HD;
#pragma unroll
    for (int t = 0; t < 4; ++t)
#pragma unroll
        for (int r = 0; r < 8; ++r)
            ob[(size_t)(r + 8 * g) * HD + t * 16 + n] = acc[t][r];
}

extern "C" void kernel_launch(void* const* d_in, const int* in_sizes, int n_in,
                              void* d_out, int out_size, void* d_ws, size_t ws_size,
                              hipStream_t stream) {
    const float* q    = (const float*)d_in[0];
    const float* k    = (const float*)d_in[1];
    const float* v    = (const float*)d_in[2];
    const int*   mask = (const int*)d_in[3];

    const int BH = in_sizes[0] / (SEQ * HD);    // B*H = 64
    float* out  = (float*)d_out;                                 // [B,H,S,D]
    float* attn = out + (size_t)BH * SEQ * HD;                   // [B,H,S,S]

    dim3 grid(SEQ / TILE_Q, BH);
    dim3 block(NTHREAD);
    ScaledDotProductAttention_28595892257161_kernel<<<grid, block, 0, stream>>>(
        q, k, v, mask, out, attn);
}
